// MiniGraphormer_42786464203512
// MI455X (gfx1250) — matrix operand order repaired and checked by reading, not compiled, not run on hardware
//
#include <hip/hip_runtime.h>
#include <math.h>

// ---------------- model constants ----------------
constexpr int BB  = 16;     // graphs
constexpr int NPG = 512;    // nodes per graph
constexpr int CC  = 128;    // channels
constexpr int HH  = 8;      // heads
constexpr int LL  = 4;      // layers
constexpr int FF  = 512;    // ffn
constexpr int DH  = 16;     // head dim
constexpr int TT  = 513;    // tokens per graph (vnode + nodes)
constexpr int NN  = BB * NPG;      // 8192 nodes
constexpr int EE  = BB * 8192;     // 131072 edges
constexpr int MM  = BB * TT;       // 8208 rows
constexpr int NQT = (TT + 15) / 16; // 33 query tiles

// ---------------- CDNA5 WMMA types ----------------
typedef __bf16 bf16_t;
typedef __attribute__((ext_vector_type(16))) __bf16 v16bf;
typedef __attribute__((ext_vector_type(8)))  __bf16 v8bf;
typedef __attribute__((ext_vector_type(8)))  float  v8f;

__device__ __forceinline__ v8f wmma_bf16(v16bf a, v16bf b, v8f c) {
  // D = A(16x32 bf16) * B(32x16 bf16) + C(16x16 f32)
  return __builtin_amdgcn_wmma_f32_16x16x32_bf16(false, a, false, b, (short)0, c,
                                                 false, false);
}

__device__ __forceinline__ float gelu_exact(float x) {
  return 0.5f * x * (1.0f + erff(x * 0.70710678118654752f));
}

// CDNA5 async global->LDS copy (16B per lane), tracked by ASYNCcnt.
__device__ __forceinline__ void async_copy_b128(uint32_t lds_byte_addr,
                                                const void* gaddr) {
  asm volatile("global_load_async_to_lds_b128 %0, %1, off"
               :
               : "v"(lds_byte_addr), "v"(gaddr)
               : "memory");
}
__device__ __forceinline__ void async_wait0() {
  asm volatile("s_wait_asynccnt 0x0" ::: "memory");
}
__device__ __forceinline__ uint32_t lds_addr32(const void* p) {
  // low 32 bits of a generic pointer into LDS == LDS byte address
  return (uint32_t)(uintptr_t)p;
}

// ---------------- degree / bias / encoder kernels ----------------
__global__ void degree_zero_kernel(int* deg) {
  int i = blockIdx.x * blockDim.x + threadIdx.x;
  if (i < 2 * NN) deg[i] = 0;
}

__global__ void degree_count_kernel(const int* __restrict__ ei,
                                    int* __restrict__ indeg,
                                    int* __restrict__ outdeg) {
  int e = blockIdx.x * blockDim.x + threadIdx.x;
  if (e >= EE) return;
  atomicAdd(&outdeg[ei[e]], 1);       // src
  atomicAdd(&indeg[ei[EE + e]], 1);   // dst
}

__global__ void bias_init_kernel(float* __restrict__ bias,
                                 const float* __restrict__ vb) {
  size_t total = (size_t)BB * HH * TT * TT;
  size_t stride = (size_t)gridDim.x * blockDim.x;
  for (size_t idx = (size_t)blockIdx.x * blockDim.x + threadIdx.x; idx < total;
       idx += stride) {
    size_t j = idx % TT;
    size_t i = (idx / TT) % TT;
    size_t h = (idx / ((size_t)TT * TT)) % HH;
    bias[idx] = (i == 0 || j == 0) ? vb[h] : 0.0f;
  }
}

__global__ void bias_edge_kernel(const int* __restrict__ ei,
                                 const int* __restrict__ ea,
                                 const float* __restrict__ ebt,
                                 float* __restrict__ bias) {
  int e = blockIdx.x * blockDim.x + threadIdx.x;
  if (e >= EE) return;
  int src = ei[e], dst = ei[EE + e];
  int a = ea[e]; a = a < 0 ? 0 : (a > 9 ? 9 : a);
  int g  = src / NPG;
  int sl = src - g * NPG + 1;
  int dl = dst - g * NPG + 1;
  for (int h = 0; h < HH; h++) {
    size_t idx = (((size_t)(g * HH + h) * TT + sl)) * TT + dl;
    atomicAdd(&bias[idx], ebt[(a + 1) * HH + h]);
  }
}

__global__ __launch_bounds__(128)
void encode_kernel(const float* __restrict__ nf, const float* __restrict__ Wn,
                   const float* __restrict__ bn, const float* __restrict__ gtok,
                   const float* __restrict__ inEmb, const float* __restrict__ outEmb,
                   const int* __restrict__ indeg, const int* __restrict__ outdeg,
                   float* __restrict__ x) {
  int r = blockIdx.x;           // 0 .. MM-1
  int c = threadIdx.x;          // 0 .. 127
  int b = r / TT, i = r % TT;
  float v;
  if (i == 0) {
    v = gtok[c];
  } else {
    int n = b * NPG + i - 1;
    v = bn[c];
    v += nf[n * 3 + 0] * Wn[0 * CC + c];
    v += nf[n * 3 + 1] * Wn[1 * CC + c];
    v += nf[n * 3 + 2] * Wn[2 * CC + c];
    int di = indeg[n];  if (di > 511) di = 511;
    int dq = outdeg[n]; if (dq > 511) dq = 511;
    v += inEmb[(size_t)di * CC + c] + outEmb[(size_t)dq * CC + c];
  }
  x[(size_t)r * CC + c] = v;
}

// transpose + convert: src f32 [layers][K][N] -> dst bf16 [layers][N][K]
__global__ void convert_transpose_kernel(const float* __restrict__ src,
                                         bf16_t* __restrict__ dst,
                                         int K, int N, int layers) {
  size_t total = (size_t)layers * K * N;
  size_t i = (size_t)blockIdx.x * blockDim.x + threadIdx.x;
  if (i >= total) return;
  size_t n = i % N;
  size_t kk = (i / N) % K;
  size_t l = i / ((size_t)K * N);
  dst[l * (size_t)K * N + n * K + kk] = (bf16_t)src[i];
}

// ---------------- LayerNorm (f32 in, bf16 out) ----------------
__global__ __launch_bounds__(128)
void ln_kernel(const float* __restrict__ x, const float* __restrict__ g,
               const float* __restrict__ bta, bf16_t* __restrict__ out) {
  __shared__ float red[8];
  int r = blockIdx.x;
  int c = threadIdx.x;
  int lane = c & 31, w = c >> 5;
  float v = x[(size_t)r * CC + c];
  float s = v;
  for (int off = 1; off < 32; off <<= 1) s += __shfl_xor(s, off);
  if (lane == 0) red[w] = s;
  __syncthreads();
  float mean = (red[0] + red[1] + red[2] + red[3]) * (1.0f / CC);
  float d = v - mean;
  float s2 = d * d;
  for (int off = 1; off < 32; off <<= 1) s2 += __shfl_xor(s2, off);
  if (lane == 0) red[4 + w] = s2;
  __syncthreads();
  float var = (red[4] + red[5] + red[6] + red[7]) * (1.0f / CC);
  float rs = rsqrtf(var + 1e-5f);
  out[(size_t)r * CC + c] = (bf16_t)(d * rs * g[c] + bta[c]);
}

// ---------------- tiled WMMA GEMM with async LDS staging ----------------
// D[M,N](f32/bf16) = A[M,K](bf16) @ BT[N,K](bf16)^T + bias[N] (+resid) (+gelu)
#define GBM 128
#define GBN 64
#define GBK 32

__global__ __launch_bounds__(256)
void gemm_bf16_kernel(const bf16_t* __restrict__ A, const bf16_t* __restrict__ BT,
                      const float* __restrict__ bias, const float* __restrict__ resid,
                      float* __restrict__ outF, bf16_t* __restrict__ outB,
                      int M, int N, int K, int do_gelu) {
  __shared__ alignas(64) bf16_t sA[GBM][GBK];
  __shared__ alignas(64) bf16_t sB[GBN][GBK];   // [n][k] layout
  int tid = threadIdx.x;
  int lane = tid & 31, wid = tid >> 5;
  int rowBase = blockIdx.y * GBM;
  int colBase = blockIdx.x * GBN;
  int half = (lane >> 4) & 1;
  int l15 = lane & 15;

  // per-thread staging coordinates
  int ar   = tid >> 1;                 // A row within tile
  int aseg = (tid & 1) * 16;           // A k-segment (16 bf16)
  int agr  = rowBase + ar;             // clamp tail rows: garbage only feeds
  if (agr > M - 1) agr = M - 1;        // output rows that are never stored
  int bn_  = tid >> 2;                 // B n-row within tile
  int bks  = (tid & 3) * 8;            // B k-segment (8 bf16)
  uint32_t ldsA0 = lds_addr32(&sA[ar][aseg]);
  uint32_t ldsA1 = lds_addr32(&sA[ar][aseg + 8]);
  uint32_t ldsB0 = lds_addr32(&sB[bn_][bks]);

  v8f acc[4];
  for (int i = 0; i < 4; i++)
    for (int j = 0; j < 8; j++) acc[i][j] = 0.0f;

  for (int k0 = 0; k0 < K; k0 += GBK) {
    // async stage A tile (128x32) and B tile (64x32) directly into LDS
    const bf16_t* ga = A + (size_t)agr * K + k0 + aseg;
    async_copy_b128(ldsA0, ga);
    async_copy_b128(ldsA1, ga + 8);
    const bf16_t* gb = BT + (size_t)(colBase + bn_) * K + k0 + bks;
    async_copy_b128(ldsB0, gb);
    async_wait0();
    __syncthreads();

    // A fragment for this wave's 16 rows (contiguous k = 16*half + i, same order as B)
    int r = wid * 16 + l15;
    v16bf af;
    {
      v8bf lo = *(const v8bf*)&sA[r][half * 16];       // k = 16*half .. +7
      v8bf hi = *(const v8bf*)&sA[r][half * 16 + 8];   // k = 16*half+8 .. +15
      for (int i = 0; i < 8; i++) { af[i] = lo[i]; af[8 + i] = hi[i]; }
    }
    for (int nt = 0; nt < 4; nt++) {
      int n = nt * 16 + l15;
      v16bf bfrag = *(const v16bf*)&sB[n][half * 16];  // k = 16*half .. +15
      acc[nt] = wmma_bf16(af, bfrag, acc[nt]);
    }
    __syncthreads();
  }

  // store (C/D layout: vgpr r8 -> row r8 + 8*half, col = lane&15)
  for (int nt = 0; nt < 4; nt++) {
    int col = colBase + nt * 16 + l15;
    float bv = bias ? bias[col] : 0.0f;
    for (int r8 = 0; r8 < 8; r8++) {
      int row = rowBase + wid * 16 + r8 + half * 8;
      if (row >= M) continue;
      float v = acc[nt][r8] + bv;
      if (resid) v += resid[(size_t)row * N + col];
      if (do_gelu) v = gelu_exact(v);
      if (outF) outF[(size_t)row * N + col] = v;
      if (outB) outB[(size_t)row * N + col] = (bf16_t)v;
    }
  }
}

// ---------------- fused flash attention ----------------
// one wave per (b, h, q-tile of 16 rows); bias streamed, softmax online.
__global__ __launch_bounds__(256)
void attention_kernel(const bf16_t* __restrict__ q, const bf16_t* __restrict__ k,
                      const bf16_t* __restrict__ v, const float* __restrict__ bias,
                      bf16_t* __restrict__ o) {
  __shared__ alignas(64) bf16_t sP[8][256];    // per-wave P tile [t][s]
  __shared__ alignas(64) bf16_t sVT[8][256];   // per-wave V^T tile [d][s]
  int lane = threadIdx.x & 31;
  int wid  = threadIdx.x >> 5;
  int job  = blockIdx.x * 8 + wid;             // BB*HH*NQT jobs
  int qt = job % NQT;
  int h  = (job / NQT) % HH;
  int b  = job / (NQT * HH);
  int half = lane >> 4;
  int l15  = lane & 15;

  // Q fragment: A(16x32), k-dim = D=16 zero-padded to 32
  v16bf aq;
  for (int i = 0; i < 16; i++) aq[i] = (bf16_t)0.0f;
  int tq = qt * 16 + l15;
  if (tq < TT && half == 0) {
    const bf16_t* p = q + ((size_t)(b * TT + tq)) * CC + h * DH;
    v16bf qv = *(const v16bf*)p;
    aq = qv;
  }

  const size_t biasRowBase = ((size_t)(b * HH + h)) * TT;

  float mrow[8], lsum[8];
  for (int r = 0; r < 8; r++) { mrow[r] = -3.0e38f; lsum[r] = 0.0f; }
  v8f oacc;
  for (int j = 0; j < 8; j++) oacc[j] = 0.0f;
  const float scale = 0.25f;  // 1/sqrt(16)

  for (int st = 0; st < NQT; st++) {
    // prefetch next s-tile of the bias stream (one row per lane 0-15)
    if (half == 0 && st + 1 < NQT && tq < TT) {
      const float* pf = &bias[(biasRowBase + tq) * TT + (st + 1) * 16];
      __builtin_prefetch(pf, 0, 0);
    }

    // K^T B-fragment: B[d][s], lanes 0-15 carry d=0..15 for col s=lane
    v16bf bk;
    for (int i = 0; i < 16; i++) bk[i] = (bf16_t)0.0f;
    if (half == 0) {
      int s = st * 16 + l15;
      if (s < TT) bk = *(const v16bf*)(k + ((size_t)(b * TT + s)) * CC + h * DH);
    }
    v8f sc;
    for (int j = 0; j < 8; j++) sc[j] = 0.0f;
    sc = wmma_bf16(aq, bk, sc);

    // stage V tile transposed into per-wave LDS: sVT[d][s]
    if (half == 0) {
      int s = st * 16 + l15;
      v16bf vv;
      for (int i = 0; i < 16; i++) vv[i] = (bf16_t)0.0f;
      if (s < TT) vv = *(const v16bf*)(v + ((size_t)(b * TT + s)) * CC + h * DH);
      for (int d = 0; d < 16; d++) sVT[wid][d * 16 + l15] = vv[d];
    }

    // scores -> scale + bias -> online softmax (row groups of 16 lanes)
    int scol = st * 16 + l15;
    for (int r = 0; r < 8; r++) {
      int m = r + half * 8;
      int trow = qt * 16 + m;
      float val;
      if (scol < TT) {
        float bb = (trow < TT) ? bias[(biasRowBase + trow) * TT + scol] : 0.0f;
        val = sc[r] * scale + bb;
      } else {
        val = -3.0e38f;
      }
      float tmax = val;
      for (int off = 1; off < 16; off <<= 1) tmax = fmaxf(tmax, __shfl_xor(tmax, off));
      float mnew  = fmaxf(mrow[r], tmax);
      float alpha = __expf(mrow[r] - mnew);
      float p     = __expf(val - mnew);
      float psum  = p;
      for (int off = 1; off < 16; off <<= 1) psum += __shfl_xor(psum, off);
      lsum[r] = lsum[r] * alpha + psum;
      mrow[r] = mnew;
      oacc[r] *= alpha;
      sP[wid][m * 16 + l15] = (bf16_t)p;
    }

    // P A-fragment (16x32, s-dim zero-padded)
    v16bf ap;
    for (int i = 0; i < 16; i++) ap[i] = (bf16_t)0.0f;
    if (half == 0) {
      v16bf pv = *(const v16bf*)&sP[wid][l15 * 16];
      ap = pv;
    }
    // V B-fragment: B[s][d], lanes 0-15 carry s=0..15 for col d=lane
    v16bf bv16;
    for (int i = 0; i < 16; i++) bv16[i] = (bf16_t)0.0f;
    if (half == 0) bv16 = *(const v16bf*)&sVT[wid][l15 * 16];

    oacc = wmma_bf16(ap, bv16, oacc);
  }

  for (int r = 0; r < 8; r++) {
    int m = r + half * 8;
    int trow = qt * 16 + m;
    if (trow < TT) {
      float inv = 1.0f / lsum[r];
      o[((size_t)(b * TT + trow)) * CC + h * DH + l15] = (bf16_t)(oacc[r] * inv);
    }
  }
}

// ---------------- final LN + classifier head ----------------
__global__ __launch_bounds__(128)
void head_kernel(const float* __restrict__ x, const float* __restrict__ gg,
                 const float* __restrict__ gb, const float* __restrict__ Wc1,
                 const float* __restrict__ bc1, const float* __restrict__ Wc2,
                 const float* __restrict__ bc2, float* __restrict__ out) {
  __shared__ float xn[CC];
  __shared__ float mid[64];
  __shared__ float red[8];
  int b = blockIdx.x, c = threadIdx.x;
  int lane = c & 31, w = c >> 5;
  float v = x[(size_t)(b * TT) * CC + c];
  float s = v;
  for (int off = 1; off < 32; off <<= 1) s += __shfl_xor(s, off);
  if (lane == 0) red[w] = s;
  __syncthreads();
  float mean = (red[0] + red[1] + red[2] + red[3]) * (1.0f / CC);
  float d = v - mean;
  float s2 = d * d;
  for (int off = 1; off < 32; off <<= 1) s2 += __shfl_xor(s2, off);
  if (lane == 0) red[4 + w] = s2;
  __syncthreads();
  float var = (red[4] + red[5] + red[6] + red[7]) * (1.0f / CC);
  xn[c] = d * rsqrtf(var + 1e-5f) * gg[c] + gb[c];
  __syncthreads();
  if (c < 64) {
    float a = bc1[c];
    for (int kk = 0; kk < CC; kk++) a += xn[kk] * Wc1[kk * 64 + c];
    mid[c] = gelu_exact(a);
  }
  __syncthreads();
  if (c < 4) {
    float a = bc2[c];
    for (int kk = 0; kk < 64; kk++) a += mid[kk] * Wc2[kk * 4 + c];
    out[b * 4 + c] = a;
  }
}

// ---------------- host orchestration ----------------
extern "C" void kernel_launch(void* const* d_in, const int* in_sizes, int n_in,
                              void* d_out, int out_size, void* d_ws, size_t ws_size,
                              hipStream_t stream) {
  const float* node_feats      = (const float*)d_in[0];
  const int*   edge_index      = (const int*)d_in[1];
  const int*   edge_attr       = (const int*)d_in[2];
  const float* W_node          = (const float*)d_in[4];
  const float* b_node          = (const float*)d_in[5];
  const float* graph_token     = (const float*)d_in[6];
  const float* edge_bias_table = (const float*)d_in[7];
  const float* in_deg_emb      = (const float*)d_in[8];
  const float* out_deg_emb     = (const float*)d_in[9];
  const float* vnode_bias      = (const float*)d_in[10];
  const float* Wq = (const float*)d_in[11];
  const float* bq = (const float*)d_in[12];
  const float* Wk = (const float*)d_in[13];
  const float* bk = (const float*)d_in[14];
  const float* Wv = (const float*)d_in[15];
  const float* bv = (const float*)d_in[16];
  const float* Wo = (const float*)d_in[17];
  const float* bo = (const float*)d_in[18];
  const float* ln1_g = (const float*)d_in[19];
  const float* ln1_b = (const float*)d_in[20];
  const float* ln2_g = (const float*)d_in[21];
  const float* ln2_b = (const float*)d_in[22];
  const float* W1 = (const float*)d_in[23];
  const float* b1 = (const float*)d_in[24];
  const float* W2 = (const float*)d_in[25];
  const float* b2 = (const float*)d_in[26];
  const float* normf_g = (const float*)d_in[27];
  const float* normf_b = (const float*)d_in[28];
  const float* Wc1 = (const float*)d_in[29];
  const float* bc1 = (const float*)d_in[30];
  const float* Wc2 = (const float*)d_in[31];
  const float* bc2 = (const float*)d_in[32];
  float* out = (float*)d_out;

  // workspace carve (~158 MB)
  char* ws = (char*)d_ws;
  size_t off = 0;
  auto carve = [&](size_t bytes) -> char* {
    char* p = ws + off;
    off += (bytes + 255) & ~(size_t)255;
    return p;
  };
  float*  biasbuf = (float*)carve((size_t)BB * HH * TT * TT * sizeof(float));
  float*  xbuf    = (float*)carve((size_t)MM * CC * sizeof(float));
  bf16_t* hbuf    = (bf16_t*)carve((size_t)MM * CC * sizeof(bf16_t));
  bf16_t* qbuf    = (bf16_t*)carve((size_t)MM * CC * sizeof(bf16_t));
  bf16_t* kbuf    = (bf16_t*)carve((size_t)MM * CC * sizeof(bf16_t));
  bf16_t* vbuf    = (bf16_t*)carve((size_t)MM * CC * sizeof(bf16_t));
  bf16_t* obuf    = (bf16_t*)carve((size_t)MM * CC * sizeof(bf16_t));
  bf16_t* midbuf  = (bf16_t*)carve((size_t)MM * FF * sizeof(bf16_t));
  bf16_t* wqT     = (bf16_t*)carve((size_t)LL * CC * CC * sizeof(bf16_t));
  bf16_t* wkT     = (bf16_t*)carve((size_t)LL * CC * CC * sizeof(bf16_t));
  bf16_t* wvT     = (bf16_t*)carve((size_t)LL * CC * CC * sizeof(bf16_t));
  bf16_t* woT     = (bf16_t*)carve((size_t)LL * CC * CC * sizeof(bf16_t));
  bf16_t* w1T     = (bf16_t*)carve((size_t)LL * CC * FF * sizeof(bf16_t));
  bf16_t* w2T     = (bf16_t*)carve((size_t)LL * FF * CC * sizeof(bf16_t));
  int*    degbuf  = (int*)carve((size_t)2 * NN * sizeof(int));
  int*    indeg   = degbuf;
  int*    outdeg  = degbuf + NN;

  // 1) degrees
  degree_zero_kernel<<<(2 * NN + 255) / 256, 256, 0, stream>>>(degbuf);
  degree_count_kernel<<<(EE + 255) / 256, 256, 0, stream>>>(edge_index, indeg, outdeg);

  // 2) attention bias tensor
  bias_init_kernel<<<4096, 256, 0, stream>>>(biasbuf, vnode_bias);
  bias_edge_kernel<<<(EE + 255) / 256, 256, 0, stream>>>(edge_index, edge_attr,
                                                         edge_bias_table, biasbuf);

  // 3) node encoder -> x
  encode_kernel<<<MM, 128, 0, stream>>>(node_feats, W_node, b_node, graph_token,
                                        in_deg_emb, out_deg_emb, indeg, outdeg, xbuf);

  // 4) weights -> bf16 transposed [N][K]
  {
    size_t t1 = (size_t)LL * CC * CC;
    size_t t2 = (size_t)LL * CC * FF;
    int g1 = (int)((t1 + 255) / 256), g2 = (int)((t2 + 255) / 256);
    convert_transpose_kernel<<<g1, 256, 0, stream>>>(Wq, wqT, CC, CC, LL);
    convert_transpose_kernel<<<g1, 256, 0, stream>>>(Wk, wkT, CC, CC, LL);
    convert_transpose_kernel<<<g1, 256, 0, stream>>>(Wv, wvT, CC, CC, LL);
    convert_transpose_kernel<<<g1, 256, 0, stream>>>(Wo, woT, CC, CC, LL);
    convert_transpose_kernel<<<g2, 256, 0, stream>>>(W1, w1T, CC, FF, LL);
    convert_transpose_kernel<<<g2, 256, 0, stream>>>(W2, w2T, FF, CC, LL);
  }

  auto gemm = [&](const bf16_t* A, const bf16_t* Bt, const float* bias_,
                  const float* resid, float* outF, bf16_t* outB, int N, int K,
                  int gelu_) {
    dim3 g(N / GBN, (MM + GBM - 1) / GBM);
    gemm_bf16_kernel<<<g, 256, 0, stream>>>(A, Bt, bias_, resid, outF, outB, MM, N,
                                            K, gelu_);
  };

  // 5) transformer layers
  for (int l = 0; l < LL; l++) {
    ln_kernel<<<MM, 128, 0, stream>>>(xbuf, ln1_g + l * CC, ln1_b + l * CC, hbuf);
    gemm(hbuf, wqT + (size_t)l * CC * CC, bq + l * CC, nullptr, nullptr, qbuf, CC, CC, 0);
    gemm(hbuf, wkT + (size_t)l * CC * CC, bk + l * CC, nullptr, nullptr, kbuf, CC, CC, 0);
    gemm(hbuf, wvT + (size_t)l * CC * CC, bv + l * CC, nullptr, nullptr, vbuf, CC, CC, 0);
    attention_kernel<<<(BB * HH * NQT) / 8, 256, 0, stream>>>(qbuf, kbuf, vbuf,
                                                              biasbuf, obuf);
    gemm(obuf, woT + (size_t)l * CC * CC, bo + l * CC, xbuf, xbuf, nullptr, CC, CC, 0);
    ln_kernel<<<MM, 128, 0, stream>>>(xbuf, ln2_g + l * CC, ln2_b + l * CC, hbuf);
    gemm(hbuf, w1T + (size_t)l * CC * FF, b1 + l * FF, nullptr, nullptr, midbuf, FF, CC, 1);
    gemm(midbuf, w2T + (size_t)l * FF * CC, b2 + l * CC, xbuf, xbuf, nullptr, CC, FF, 0);
  }

  // 6) final LN + head on graph tokens
  head_kernel<<<BB, 128, 0, stream>>>(xbuf, normf_g, normf_b, Wc1, bc1, Wc2, bc2, out);
}
